// VectorQuantizer_52570399703653
// MI455X (gfx1250) — compile-verified, matrix operand order checked by reading, not run on hardware
//
#include <hip/hip_runtime.h>

typedef __attribute__((ext_vector_type(16))) _Float16 v16h;
typedef __attribute__((ext_vector_type(8)))  _Float16 v8h;
typedef __attribute__((ext_vector_type(8)))  float    v8f;
typedef __attribute__((ext_vector_type(4)))  float    v4f;

#define NUM_EMB        512
#define EMB_DIM        64
#define HALF_K         256          // codebook rows resident in LDS per phase
#define WPAD           72           // padded LDS row stride in halves (144B): every b128 is 2-clk minimum
#define BLOCK          256          // 8 waves (wave32)
#define ROWS_PER_WAVE  32           // two 16-row A tiles share each B fragment
#define ROWS_PER_BLOCK 256
#define NTILES         (HALF_K / 16)
#define IDX_MASK       0x1FFu       // 9-bit code index in low mantissa bits
#define KEY_MASK       0xFFFFFE00u

__device__ __forceinline__ void load_b(const _Float16* __restrict__ sW, int nb, int m, int ks,
                                       v16h& b0, v16h& b1) {
  const _Float16* wrow = &sW[(nb + m) * WPAD];
  v8h l0 = *(const v8h*)(wrow + ks);
  v8h h0 = *(const v8h*)(wrow + ks + 16);
  v8h l1 = *(const v8h*)(wrow + 32 + ks);
  v8h h1 = *(const v8h*)(wrow + 32 + ks + 16);
  b0 = __builtin_shufflevector(l0, h0, 0,1,2,3,4,5,6,7,8,9,10,11,12,13,14,15);
  b1 = __builtin_shufflevector(l1, h1, 0,1,2,3,4,5,6,7,8,9,10,11,12,13,14,15);
}

__device__ __forceinline__ void load_a(const float* __restrict__ zrow, int ks,
                                       v16h& a0, v16h& a1, float& zsq) {
  v8f zl0 = *(const v8f*)(zrow + ks);
  v8f zh0 = *(const v8f*)(zrow + ks + 16);
  v8f zl1 = *(const v8f*)(zrow + 32 + ks);
  v8f zh1 = *(const v8f*)(zrow + 32 + ks + 16);
  float s = 0.f;
  #pragma unroll
  for (int i = 0; i < 8; ++i) {
    a0[i]     = (_Float16)zl0[i];
    a0[8 + i] = (_Float16)zh0[i];
    a1[i]     = (_Float16)zl1[i];
    a1[8 + i] = (_Float16)zh1[i];
    s = __builtin_fmaf(zl0[i], zl0[i], s);
    s = __builtin_fmaf(zh0[i], zh0[i], s);
    s = __builtin_fmaf(zl1[i], zl1[i], s);
    s = __builtin_fmaf(zh1[i], zh1[i], s);
  }
  zsq = s;
}

__global__ __launch_bounds__(BLOCK) void vq_argmin_gather_kernel(
    const float* __restrict__ z, const float* __restrict__ W,
    float* __restrict__ zq_out, float* __restrict__ idx_out,
    float* __restrict__ partial)
{
  __shared__ _Float16 sW[HALF_K * WPAD];      // 36864 B
  __shared__ float    sWn[HALF_K];            // 1024 B : ||W_k||^2 + 4.0  (positivity bias)
  __shared__ int      sIdx[ROWS_PER_BLOCK];   // 1024 B
  __shared__ float    sLoss[ROWS_PER_BLOCK];  // 1024 B

  const int tid  = threadIdx.x;
  const int lane = tid & 31;
  const int wave = tid >> 5;
  const int m    = lane & 15;       // A-row within tile / C column
  const int hi   = lane >> 4;       // K-half selector for A/B fragments
  const int ks   = hi * 8;
  const int rowbase = blockIdx.x * ROWS_PER_BLOCK + wave * ROWS_PER_WAVE;

  // ---- A fragments for two 16-row sets, plus f32 row-norm partials ----
  const float* zrow0 = z + (size_t)(rowbase + m) * EMB_DIM;
  const float* zrow1 = zrow0 + (size_t)16 * EMB_DIM;
  v16h a0, a1, a2, a3;
  float zsq0, zsq1;
  load_a(zrow0, ks, a0, a1, zsq0);
  load_a(zrow1, ks, a2, a3, zsq1);
  zsq0 += __shfl_xor(zsq0, 16, 32);   // combine disjoint half-rows -> full ||z_row||^2
  zsq1 += __shfl_xor(zsq1, 16, 32);

  // running argmin as packed keys: [positive-score bits 31:9 | code index 8:0]
  unsigned int best0[8], best1[8];
  #pragma unroll
  for (int j = 0; j < 8; ++j) { best0[j] = 0xFFFFFFFFu; best1[j] = 0xFFFFFFFFu; }

  for (int phase = 0; phase < 2; ++phase) {
    const int kbase = phase * HALF_K;
    // ---- stage 256 codebook rows into LDS as f16, biased norms in f32 ----
    {
      const float* wr = W + (size_t)(kbase + tid) * EMB_DIM;
      _Float16* dst = &sW[tid * WPAD];
      float nrm = 0.f;
      #pragma unroll
      for (int c = 0; c < EMB_DIM; ++c) {
        float v = wr[c];
        nrm = __builtin_fmaf(v, v, nrm);
        dst[c] = (_Float16)v;
      }
      sWn[tid] = nrm + 4.0f;    // score s' = (||W||^2+4) - 2 z.W  in [3.5, 4.5] > 0
    }
    __syncthreads();

    v16h b0, b1, p0, p1;
    load_b(sW, 0, m, ks, b0, b1);
    #pragma unroll 2
    for (int t = 0; t < NTILES; ++t) {
      load_b(sW, ((t + 1) & (NTILES - 1)) * 16, m, ks, p0, p1);  // prefetch next (wraps harmlessly)

      v8f c0 = {}, c1 = {};
      c0 = __builtin_amdgcn_wmma_f32_16x16x32_f16(false, a0, false, b0, (short)0, c0, false, false);
      c0 = __builtin_amdgcn_wmma_f32_16x16x32_f16(false, a1, false, b1, (short)0, c0, false, false);
      c1 = __builtin_amdgcn_wmma_f32_16x16x32_f16(false, a2, false, b0, (short)0, c1, false, false);
      c1 = __builtin_amdgcn_wmma_f32_16x16x32_f16(false, a3, false, b1, (short)0, c1, false, false);

      const float        wn   = sWn[t * 16 + m];               // ds_load_b32, overlapped
      const unsigned int colg = (unsigned int)(kbase + t * 16 + m);
      #pragma unroll
      for (int j = 0; j < 8; ++j) {
        float s0 = __builtin_fmaf(-2.0f, c0[j], wn);   // positive score
        float s1 = __builtin_fmaf(-2.0f, c1[j], wn);
        unsigned int k0 = (__float_as_uint(s0) & KEY_MASK) | (colg & IDX_MASK);  // v_bfi
        unsigned int k1 = (__float_as_uint(s1) & KEY_MASK) | (colg & IDX_MASK);
        best0[j] = min(best0[j], k0);                  // v_min_u32: score then lowest idx
        best1[j] = min(best1[j], k1);
      }
      b0 = p0; b1 = p1;
    }
    __syncthreads();   // all waves done reading sW before next phase overwrites
  }

  // ---- cross-lane argmin: pure u32 min of packed keys ----
  #pragma unroll
  for (int mask = 8; mask >= 1; mask >>= 1) {
    #pragma unroll
    for (int j = 0; j < 8; ++j) {
      best0[j] = min(best0[j], (unsigned int)__shfl_xor((int)best0[j], mask, 32));
      best1[j] = min(best1[j], (unsigned int)__shfl_xor((int)best1[j], mask, 32));
    }
  }

  // indices: lanes 0 (rows 0..7) / 16 (rows 8..15) of each group
  if (m == 0) {
    #pragma unroll
    for (int j = 0; j < 8; ++j) {
      sIdx[wave * ROWS_PER_WAVE + hi * 8 + j]      = (int)(best0[j] & IDX_MASK);
      sIdx[wave * ROWS_PER_WAVE + 16 + hi * 8 + j] = (int)(best1[j] & IDX_MASK);
    }
  }

  // per-row loss: ||z||^2 + (score - 4). Owner lanes: 0..7 (rows 0..7), 24..31 (rows 8..15)
  {
    unsigned int k0 = 0, k1 = 0;
    const int j7 = lane & 7;
    #pragma unroll
    for (int j = 0; j < 8; ++j) {
      bool sel = (j7 == j);
      k0 = sel ? best0[j] : k0;
      k1 = sel ? best1[j] : k1;
    }
    float s0 = __uint_as_float(k0 & KEY_MASK) - 4.0f;
    float s1 = __uint_as_float(k1 & KEY_MASK) - 4.0f;
    bool owner = (lane < 8) | (lane >= 24);
    int  rloc  = j7 + hi * 8;
    if (owner) {
      sLoss[wave * ROWS_PER_WAVE + rloc]      = zsq0 + s0;
      sLoss[wave * ROWS_PER_WAVE + 16 + rloc] = zsq1 + s1;
    }
  }
  __syncthreads();

  // ---- write indices as float ----
  idx_out[blockIdx.x * ROWS_PER_BLOCK + tid] = (float)sIdx[tid];

  // ---- coalesced gather of f32 codebook rows (L2-resident) -> z_q ----
  {
    const v4f* W4 = (const v4f*)W;
    v4f* out4 = (v4f*)(zq_out + (size_t)blockIdx.x * (ROWS_PER_BLOCK * EMB_DIM));
    #pragma unroll
    for (int i = 0; i < (ROWS_PER_BLOCK * EMB_DIM / 4) / BLOCK; ++i) {  // 16 iters
      int e4 = i * BLOCK + tid;
      int r  = e4 >> 4;                 // 16 quads per row
      int c4 = e4 & 15;
      int idx = sIdx[r];
      out4[e4] = W4[idx * (EMB_DIM / 4) + c4];
    }
  }

  // ---- deterministic in-place tree reduction of per-row losses ----
  for (int s = ROWS_PER_BLOCK / 2; s > 0; s >>= 1) {
    if (tid < s) sLoss[tid] += sLoss[tid + s];
    __syncthreads();
  }
  if (tid == 0) partial[blockIdx.x] = sLoss[0];
}

__global__ __launch_bounds__(256) void vq_finalize_kernel(
    const float* __restrict__ partial, int nblocks,
    float* __restrict__ loss_out, float inv_total)
{
  __shared__ float sR[256];
  float a = 0.f;
  for (int i = threadIdx.x; i < nblocks; i += 256) a += partial[i];
  sR[threadIdx.x] = a;
  __syncthreads();
  for (int s = 128; s > 0; s >>= 1) {
    if (threadIdx.x < s) sR[threadIdx.x] += sR[threadIdx.x + s];
    __syncthreads();
  }
  if (threadIdx.x == 0)
    loss_out[0] = 1.25f * sR[0] * inv_total;   // mse + 0.25*mse (sg is value no-op)
}

extern "C" void kernel_launch(void* const* d_in, const int* in_sizes, int n_in,
                              void* d_out, int out_size, void* d_ws, size_t ws_size,
                              hipStream_t stream) {
  const float* z = (const float*)d_in[0];
  const float* W = (const float*)d_in[1];
  const int nz    = in_sizes[0];             // 8*32768*64 = 16777216
  const int nrows = nz / EMB_DIM;            // 262144
  const int nblk  = nrows / ROWS_PER_BLOCK;  // 1024

  float* out   = (float*)d_out;
  float* zq    = out;                        // [nz]    z_q_st (f32)
  float* loss  = out + nz;                   // [1]     loss
  float* idxf  = out + nz + 1;               // [nrows] indices as f32
  float* parts = (float*)d_ws;               // [nblk]  per-block partial sums

  vq_argmin_gather_kernel<<<nblk, BLOCK, 0, stream>>>(z, W, zq, idxf, parts);
  vq_finalize_kernel<<<1, 256, 0, stream>>>(parts, nblk, loss, 1.0f / (float)nz);
}
